// CorrelationLayer_61933428417187
// MI455X (gfx1250) — matrix operand order repaired and checked by reading, not compiled, not run on hardware
//
#include <hip/hip_runtime.h>

// ---------------------------------------------------------------------------
// Correlation volume (FlowNet-style) for MI455X / gfx1250.
//   out[b, (di+20)*41+(dj+20), h, w] = (1/C) * sum_c x1[b,c,h,w]*x2[b,c,h+di,w+dj]
// Banded Gram-matrix GEMM per (b,h,di): S[w,w'] = x1row[w].x2row[w'] via
// v_wmma_f32_16x16x32_bf16; x2 row staged in LDS once per block; fragments
// preloaded per K-step so WMMAs issue back-to-back.
// ---------------------------------------------------------------------------

typedef __bf16 bf16;
typedef __attribute__((ext_vector_type(16))) __bf16 v16bf;
typedef __attribute__((ext_vector_type(8)))  __bf16 v8bf;
typedef __attribute__((ext_vector_type(8)))  float  v8f;

#define BB   4
#define CC   128
#define HH   64
#define WW   64
#define DD   41      // displacements per axis
#define HP   104     // 64 + 2*20 padded rows
#define WP   128     // 64 + 32 left + 32 right padded cols
#define PADH 20
#define PADW 32
#define BSTR 136     // LDS col stride in bf16 (128 + 8 pad -> 272B = 17 x 16B)

// ---- prepass: x1 [B,C,H,W] f32 -> A [B,H,W,C] bf16 (coalesced writes) ------
__global__ __launch_bounds__(256) void cvt_x1_kernel(const float* __restrict__ x,
                                                     bf16* __restrict__ A) {
    int idx = blockIdx.x * 256 + threadIdx.x;       // B*H*W*C = 2^21 exactly
    int c = idx & (CC - 1);
    int w = (idx >> 7) & (WW - 1);
    int h = (idx >> 13) & (HH - 1);
    int b = idx >> 19;
    A[idx] = (bf16)x[(((size_t)b * CC + c) * HH + h) * WW + w];
}

// ---- prepass: x2 [B,C,H,W] f32 -> Bm [B,HP,WP,C] bf16, zero padded ---------
__global__ __launch_bounds__(256) void cvt_x2_kernel(const float* __restrict__ x,
                                                     bf16* __restrict__ Bm) {
    int idx = blockIdx.x * 256 + threadIdx.x;       // B*HP*WP*C = 6,815,744 exactly
    int c    = idx & (CC - 1);
    int wp   = (idx >> 7) & (WP - 1);
    int rest = idx >> 14;                           // b*HP + hp
    int hp   = rest % HP;
    int b    = rest / HP;
    int h = hp - PADH;
    int w = wp - PADW;
    float v = 0.0f;
    if ((unsigned)h < (unsigned)HH && (unsigned)w < (unsigned)WW)
        v = x[(((size_t)b * CC + c) * HH + h) * WW + w];
    Bm[idx] = (bf16)v;
}

// ---- main: banded Gram GEMM + band scatter ---------------------------------
// grid.x = B*H*41 ; block = 128 threads = 4 waves, wave handles w-tile wt.
__global__ __launch_bounds__(128) void corr_wmma_kernel(const bf16* __restrict__ A,
                                                        const bf16* __restrict__ Bm,
                                                        float* __restrict__ out) {
    __shared__ bf16  Bs[WP * BSTR];                 // staged x2 row, 34816 B
    __shared__ float S[4][16 * 80];                 // per-wave S tile, 20480 B

    const int tid  = threadIdx.x;
    const int lane = tid & 31;
    const int wt   = tid >> 5;                      // 0..3 (16-wide w tile)
    int bid = blockIdx.x;
    const int di = bid % DD;                        // 0..40 vertical disp index
    bid /= DD;
    const int h = bid % HH;
    const int b = bid / HH;

    const int m    = lane & 15;                     // A row / B col within tile
    const int half = lane >> 4;                     // K-half select

    // ---- cooperative stage of x2 padded row (b, h+di): 32KB contiguous ----
    // 2048 chunks of 16B; consecutive lanes read consecutive chunks (coalesced).
    {
        const bf16* gsrc = Bm + ((size_t)(b * HP + h + di) * WP) * CC;
        #pragma unroll
        for (int i = 0; i < 16; ++i) {
            int c16 = tid + i * 128;                // chunk id, 8 bf16 each
            int col = c16 >> 4;                     // 16 chunks per column
            int off = (c16 & 15) * 8;               // bf16 offset within column
            v8bf v = *(const v8bf*)(gsrc + c16 * 8);
            *(v8bf*)(&Bs[col * BSTR + off]) = v;
        }
    }
    __syncthreads();

    // A row pointer: w = wt*16 + m, layout [B,H,W,C]
    const bf16* arow = A + (((size_t)(b * HH + h) * WW + wt * 16 + m) * CC);

    v8f acc[5] = {};                                // five 16x16 f32 tiles

    #pragma unroll
    for (int k = 0; k < 4; ++k) {                   // K = 128 in chunks of 32
        const int kb = k * 32;
        // A fragment (16x32 bf16): lanes 0-15 rows M, K {0..7,16..23};
        // lanes 16-31 same rows, K {8..15,24..31}.
        v8bf alo = *(const v8bf*)(arow + kb + half * 8);
        v8bf ahi = *(const v8bf*)(arow + kb + 16 + half * 8);
        v16bf afrag;
        #pragma unroll
        for (int i = 0; i < 8; ++i) { afrag[i] = alo[i]; afrag[i + 8] = ahi[i]; }

        // Preload all five B fragments from LDS (lane n = column, K {8h..8h+7, 16+8h..16+8h+7}).
        v16bf bfrag[5];
        #pragma unroll
        for (int nt = 0; nt < 5; ++nt) {
            const bf16* bcol = &Bs[(wt * 16 + nt * 16 + m) * BSTR + kb + half * 8];
            v8bf blo = *(const v8bf*)(bcol);
            v8bf bhi = *(const v8bf*)(bcol + 16);
            #pragma unroll
            for (int i = 0; i < 8; ++i) { bfrag[nt][i] = blo[i]; bfrag[nt][i + 8] = bhi[i]; }
        }

        // Back-to-back WMMA chain for this K-step.
        #pragma unroll
        for (int nt = 0; nt < 5; ++nt)
            acc[nt] = __builtin_amdgcn_wmma_f32_16x16x32_bf16(
                false, afrag, false, bfrag[nt], (short)0, acc[nt], false, false);
    }

    // Stage S tile in LDS: C/D layout -> VGPR r holds (M = r + 8*half, N = lane&15)
    #pragma unroll
    for (int nt = 0; nt < 5; ++nt)
        #pragma unroll
        for (int r = 0; r < 8; ++r)
            S[wt][(r + 8 * half) * 80 + nt * 16 + m] = acc[nt][r];

    // Scatter the valid band, coalesced per channel:
    // lane j (0..15) handles w = wt*16 + j; halves cover two dj per iteration.
    const size_t hw = (size_t)h * WW + wt * 16 + m;
    #pragma unroll
    for (int p = 0; p < 21; ++p) {
        int dj = p * 2 + half;                      // dj index 0..41
        if (dj < DD) {
            int col = m + dj + 12;                  // w' - (wt*16 - 32), in [12,67]
            float v = S[wt][m * 80 + col] * 0.0078125f;   // mean over C=128
            size_t ch = (size_t)di * DD + dj;
            out[((size_t)b * (DD * DD) + ch) * (HH * WW) + hw] = v;
        }
    }
}

extern "C" void kernel_launch(void* const* d_in, const int* in_sizes, int n_in,
                              void* d_out, int out_size, void* d_ws, size_t ws_size,
                              hipStream_t stream) {
    (void)in_sizes; (void)n_in; (void)out_size; (void)ws_size;
    const float* x1 = (const float*)d_in[0];
    const float* x2 = (const float*)d_in[1];
    float* out = (float*)d_out;

    bf16* A  = (bf16*)d_ws;                               // 4*64*64*128*2  = 4.19 MB
    bf16* Bm = A + (size_t)BB * HH * WW * CC;             // 4*104*128*128*2 = 13.6 MB

    cvt_x1_kernel<<<(BB * CC * HH * WW) / 256, 256, 0, stream>>>(x1, A);
    cvt_x2_kernel<<<(BB * HP * WP * CC) / 256, 256, 0, stream>>>(x2, Bm);
    corr_wmma_kernel<<<BB * HH * DD, 128, 0, stream>>>(A, Bm, out);
}
